// Linear_int8_T_50483045597929
// MI455X (gfx1250) — 8-bit matrix operand order repaired and checked by reading, compiled, not run on hardware
//
#include <hip/hip_runtime.h>
#include <stdint.h>

typedef int v8i __attribute__((ext_vector_type(8)));

__global__ void init_ws_kernel(unsigned* bits) {
    if (threadIdx.x < 2) bits[threadIdx.x] = 0u;
}

__global__ void absmax_kernel(const float* __restrict__ p, int n,
                              unsigned* __restrict__ outbits) {
    int i      = blockIdx.x * blockDim.x + threadIdx.x;
    int stride = gridDim.x * blockDim.x;
    float m = 0.0f;
    for (; i < n; i += stride) m = fmaxf(m, fabsf(p[i]));
    #pragma unroll
    for (int off = 16; off > 0; off >>= 1)
        m = fmaxf(m, __shfl_xor(m, off, 32));
    if ((threadIdx.x & 31) == 0) atomicMax(outbits, __float_as_uint(m));
}

__global__ void quant_kernel(const float4* __restrict__ in, int n4,
                             const float* __restrict__ T_ema,
                             const unsigned* __restrict__ maxbits,
                             int* __restrict__ outq) {
    int i = blockIdx.x * blockDim.x + threadIdx.x;
    if (i >= n4) return;
    float T  = 0.95f * T_ema[0] + 0.05f * __uint_as_float(maxbits[0]);
    float rs = 127.0f / T;
    float4 v = in[i];
    int q0 = min(127, max(-128, __float2int_rn(v.x * rs)));
    int q1 = min(127, max(-128, __float2int_rn(v.y * rs)));
    int q2 = min(127, max(-128, __float2int_rn(v.z * rs)));
    int q3 = min(127, max(-128, __float2int_rn(v.w * rs)));
    outq[i] = (q0 & 255) | ((q1 & 255) << 8) | ((q2 & 255) << 16) | ((q3 & 255) << 24);
}

__global__ __launch_bounds__(256)
void gemm_iu8_kernel(const int8_t* __restrict__ qx,
                     const int8_t* __restrict__ qw,
                     const float*  __restrict__ bias,
                     const float*  __restrict__ Tf0,
                     const float*  __restrict__ Tw0,
                     const unsigned* __restrict__ maxbits,
                     float* __restrict__ out) {
    const int lane = threadIdx.x & 31;
    const int wave = threadIdx.x >> 5;
    const int l15  = lane & 15;
    const int hl   = lane >> 4;

    const int mt = blockIdx.x >> 3;
    const int nt = ((blockIdx.x & 7) << 3) + wave;

    const int8_t* arow = qx + (size_t)(mt * 16 + l15) * 1024;
    const int8_t* brow = qw + (size_t)(nt * 16 + l15) * 1024;

    v8i acc = {};
    #pragma unroll 4
    for (int kk = 0; kk < 1024; kk += 64) {
        v8i a, b;
        #pragma unroll
        for (int j = 0; j < 8; ++j) {
            const int ka = kk + (hl << 3) + ((j & 1) << 2) + (((j >> 1) & 1) << 4) + ((j >> 2) << 5);
            const int kb = kk + (hl << 3) + ((j & 1) << 2) + (((j >> 1) & 1) << 4) + ((j >> 2) << 5);
            a[j] = *(const int*)(arow + ka);
            b[j] = *(const int*)(brow + kb);
        }
        acc = __builtin_amdgcn_wmma_i32_16x16x64_iu8(true, a, true, b, acc,
                                                     false, false);
    }

    const float Tf = 0.95f * Tf0[0] + 0.05f * __uint_as_float(maxbits[0]);
    const float Tw = 0.95f * Tw0[0] + 0.05f * __uint_as_float(maxbits[1]);
    const float scale = (Tf / 127.0f) * (Tw / 127.0f);

    const int n     = nt * 16 + l15;
    const int mbase = mt * 16 + hl * 8;
    const float bi  = bias[n];
    #pragma unroll
    for (int r = 0; r < 8; ++r)
        out[(size_t)(mbase + r) * 1024 + n] = (float)acc[r] * scale + bi;
}

extern "C" void kernel_launch(void* const* d_in, const int* in_sizes, int n_in,
                              void* d_out, int out_size, void* d_ws, size_t ws_size,
                              hipStream_t stream) {
    const float* x      = (const float*)d_in[0];
    const float* weight = (const float*)d_in[1];
    const float* bias   = (const float*)d_in[2];
    const float* Tf     = (const float*)d_in[4];
    const float* Tw     = (const float*)d_in[5];
    float* out          = (float*)d_out;

    unsigned* bits = (unsigned*)d_ws;
    int8_t* qx = (int8_t*)d_ws + 256;
    int8_t* qw = qx + 128 * 1024;

    const int NX = 128 * 1024;
    const int NW = 1024 * 1024;

    init_ws_kernel<<<1, 32, 0, stream>>>(bits);
    absmax_kernel<<<128, 256, 0, stream>>>(x, NX, bits);
    absmax_kernel<<<512, 256, 0, stream>>>(weight, NW, bits + 1);
    quant_kernel<<<(NX / 4 + 255) / 256, 256, 0, stream>>>(
        (const float4*)x, NX / 4, Tf, bits, (int*)qx);
    quant_kernel<<<(NW / 4 + 255) / 256, 256, 0, stream>>>(
        (const float4*)weight, NW / 4, Tw, bits + 1, (int*)qw);
    gemm_iu8_kernel<<<64, 256, 0, stream>>>(qx, qw, bias, Tf, Tw, bits, out);
}
